// EdgeWeightedGNN_3332894622502
// MI455X (gfx1250) — matrix operand order repaired and checked by reading, not compiled, not run on hardware
//
#include <hip/hip_runtime.h>
#include <hip/hip_bf16.h>

#define N_NODES 40000
#define N_EDGES 640000
#define HDIM    128
#define NTILE   (N_EDGES / 16)
#define NEG_KEY 0x007FFFFFu   // fkey(-inf)

typedef __attribute__((ext_vector_type(16))) __bf16 v16bf;
typedef __attribute__((ext_vector_type(8)))  float  v8f;
typedef __attribute__((ext_vector_type(4)))  unsigned int u32x4;
typedef __attribute__((ext_vector_type(4)))  float  f32x4;

// ---- monotonic float <-> ordered-u32 key (for atomic max) ----
__device__ __forceinline__ unsigned fkey(float f) {
    unsigned u = __float_as_uint(f);
    return (u & 0x80000000u) ? ~u : (u | 0x80000000u);
}
__device__ __forceinline__ float fdec(unsigned k) {
    unsigned u = (k & 0x80000000u) ? (k & 0x7FFFFFFFu) : ~k;
    return __uint_as_float(u);
}
__device__ __forceinline__ unsigned pk2bf(float a, float b) {
    union { __bf16 h[2]; unsigned u; } p;
    p.h[0] = (__bf16)a; p.h[1] = (__bf16)b;
    return p.u;
}
__device__ __forceinline__ v16bf load_frag(const __bf16* p) { // p 16B-aligned
    union { u32x4 q[2]; v16bf v; } u;
    u.q[0] = *(const u32x4*)p;
    u.q[1] = *(const u32x4*)(p + 8);
    return u.v;
}

// ---- pack W1 (L,256,128) / W2 (L,128,128) fp32 -> bf16 A-fragments ----
// A = W^T : A[m][k] = W[k][m].  Fragment layout (16-bit A 16x32, this code's contiguous k order):
// lane l: row m = 16*mt + (l&15); elem e: k = 32*kt + 16*hi + e
__global__ void pack_w_kernel(const float* __restrict__ W1,
                              const float* __restrict__ W2,
                              __bf16* __restrict__ pW1, __bf16* __restrict__ pW2) {
    int idx = blockIdx.x * 256 + threadIdx.x;
    if (idx < 3 * 8 * 8 * 512) { // W1: [i][mt][kt][lane][e]
        int e = idx & 15, l = (idx >> 4) & 31, kt = (idx >> 9) & 7;
        int mt = (idx >> 12) & 7, i = idx >> 15;
        int hi = l >> 4, m = 16 * mt + (l & 15);
        int kk = 16 * hi + e;
        int k = 32 * kt + kk;
        pW1[idx] = (__bf16)W1[(i * 256 + k) * 128 + m];
    }
    int j = idx - 3 * 8 * 8 * 512;
    if (j >= 0 && j < 3 * 8 * 4 * 512) { // W2: [i][mt][kt][lane][e]
        int e = j & 15, l = (j >> 4) & 31, kt = (j >> 9) & 3;
        int mt = (j >> 11) & 7, i = j >> 14;
        int hi = l >> 4, m = 16 * mt + (l & 15);
        int kk = 16 * hi + e;
        int k = 32 * kt + kk;
        pW2[j] = (__bf16)W2[(i * 128 + k) * 128 + m];
    }
}

// ---- h = relu(x @ Wp + bp) : tiny (0.66 GFLOP), scalar is fine ----
__global__ void proj_kernel(const float* __restrict__ x, const float* __restrict__ Wp,
                            const float* __restrict__ bp, float* __restrict__ h) {
    int i = blockIdx.x * 256 + threadIdx.x;
    if (i >= N_NODES * HDIM) return;
    int n = i >> 7, jf = i & 127;
    float acc = bp[jf];
    const float* xr = x + (size_t)n * 64;
#pragma unroll 8
    for (int k = 0; k < 64; ++k) acc = fmaf(xr[k], Wp[k * 128 + jf], acc);
    h[i] = fmaxf(acc, 0.f);
}

__global__ void init_kernel(unsigned* __restrict__ keys, float* __restrict__ stats) {
    int i = blockIdx.x * 256 + threadIdx.x;
    if (i < N_NODES * HDIM) keys[i] = NEG_KEY;
    if (i < 256) stats[i] = 0.f;
}

// ---- edge MLP, WMMA bf16: D = (W^T)(m^T); waves own 16-feature slices ----
__global__ __launch_bounds__(256) void edge_mlp_kernel(
    const float* __restrict__ h, const int* __restrict__ ei,
    const __bf16* __restrict__ pW1, const __bf16* __restrict__ pW2,
    const float* __restrict__ b1, const float* __restrict__ b2,
    unsigned* __restrict__ keys, int layer) {
    __shared__ __align__(16) __bf16 m_lds[16][264]; // [edge][2H feat], padded
    __shared__ __align__(16) __bf16 t_lds[16][136]; // [edge][H feat], padded

    const int tid = threadIdx.x;
    const int lane = tid & 31;
    const int wv   = tid >> 5;     // wave -> output-feature tile (0..7)
    const int col  = lane & 15;    // B/D column = edge within tile
    const int hi   = lane >> 4;

    // Per-wave weight fragments: resident for the whole kernel.
    v16bf w1f[8], w2f[4];
    {
        const __bf16* b1p = pW1 + (size_t)((layer * 8 + wv) * 8) * 512 + lane * 16;
#pragma unroll
        for (int kt = 0; kt < 8; ++kt) w1f[kt] = load_frag(b1p + kt * 512);
        const __bf16* b2p = pW2 + (size_t)((layer * 8 + wv) * 4) * 512 + lane * 16;
#pragma unroll
        for (int kt = 0; kt < 4; ++kt) w2f[kt] = load_frag(b2p + kt * 512);
    }
    v8f bias1, bias2; // D rows m = 16*wv + 8*hi + r
#pragma unroll
    for (int r = 0; r < 8; ++r) {
        bias1[r] = b1[layer * 128 + 16 * wv + 8 * hi + r];
        bias2[r] = b2[layer * 128 + 16 * wv + 8 * hi + r];
    }

    const int ge = tid >> 4;          // gather: edge 0..15
    const int gf = (tid & 15) * 8;    // gather: feature base 0..120

    for (int tile = blockIdx.x; tile < NTILE; tile += gridDim.x) {
        const int e0 = tile * 16;
        // ---- gather m = [xi | xj - xi] into LDS (bf16) ----
        {
            int eg = e0 + ge;
            int s = ei[eg];               // src -> x_j
            int d = ei[N_EDGES + eg];     // dst -> x_i
            const float* xip = h + (size_t)d * 128 + gf;
            const float* xjp = h + (size_t)s * 128 + gf;
            f32x4 a0 = *(const f32x4*)xip,       a1 = *(const f32x4*)(xip + 4);
            f32x4 c0 = *(const f32x4*)xjp,       c1 = *(const f32x4*)(xjp + 4);
            u32x4 mi = { pk2bf(a0[0], a0[1]), pk2bf(a0[2], a0[3]),
                         pk2bf(a1[0], a1[1]), pk2bf(a1[2], a1[3]) };
            u32x4 md = { pk2bf(c0[0]-a0[0], c0[1]-a0[1]), pk2bf(c0[2]-a0[2], c0[3]-a0[3]),
                         pk2bf(c1[0]-a1[0], c1[1]-a1[1]), pk2bf(c1[2]-a1[2], c1[3]-a1[3]) };
            *(u32x4*)&m_lds[ge][gf]       = mi;
            *(u32x4*)&m_lds[ge][128 + gf] = md;
        }
        __syncthreads();

        // ---- GEMM1: T(16 feats x 16 edges) = W1^T . m^T, K=256 ----
        v8f acc = bias1;
#pragma unroll
        for (int kt = 0; kt < 8; ++kt) {
            v16bf bf = load_frag(&m_lds[col][32 * kt + 16 * hi]);
            acc = __builtin_amdgcn_wmma_f32_16x16x32_bf16(
                false, w1f[kt], false, bf, (short)0, acc, false, false);
        }
        // relu + write T to LDS in [edge][feature] bf16
        {
            u32x4 tp = { pk2bf(fmaxf(acc[0],0.f), fmaxf(acc[1],0.f)),
                         pk2bf(fmaxf(acc[2],0.f), fmaxf(acc[3],0.f)),
                         pk2bf(fmaxf(acc[4],0.f), fmaxf(acc[5],0.f)),
                         pk2bf(fmaxf(acc[6],0.f), fmaxf(acc[7],0.f)) };
            *(u32x4*)&t_lds[col][16 * wv + 8 * hi] = tp;
        }
        __syncthreads();

        // ---- GEMM2: out = W2^T . relu(T), K=128 ----
        v8f acc2 = bias2;
#pragma unroll
        for (int kt = 0; kt < 4; ++kt) {
            v16bf bf = load_frag(&t_lds[col][32 * kt + 16 * hi]);
            acc2 = __builtin_amdgcn_wmma_f32_16x16x32_bf16(
                false, w2f[kt], false, bf, (short)0, acc2, false, false);
        }
        // ---- scatter-max into agg via ordered-key u32 atomics ----
        {
            int eg = e0 + col;
            int node = ei[N_EDGES + eg]; // dst
            unsigned* rp = keys + (size_t)node * 128 + 16 * wv + 8 * hi;
#pragma unroll
            for (int r = 0; r < 8; ++r) atomicMax(rp + r, fkey(acc2[r]));
        }
        __syncthreads();
    }
}

// ---- per-feature sum / sumsq, one block per feature (deterministic) ----
__global__ __launch_bounds__(256) void stats_kernel(const unsigned* __restrict__ keys,
                                                    float* __restrict__ stats) {
    int f = blockIdx.x; // 0..127
    float s = 0.f, s2 = 0.f;
    for (int n = threadIdx.x; n < N_NODES; n += 256) {
        unsigned k = keys[(size_t)n * 128 + f];
        float v = (k == NEG_KEY) ? 0.f : fdec(k);
        s += v; s2 += v * v;
    }
    __shared__ float sh[256], sh2[256];
    sh[threadIdx.x] = s; sh2[threadIdx.x] = s2;
    __syncthreads();
    for (int st = 128; st > 0; st >>= 1) {
        if (threadIdx.x < st) { sh[threadIdx.x] += sh[threadIdx.x + st];
                                sh2[threadIdx.x] += sh2[threadIdx.x + st]; }
        __syncthreads();
    }
    if (threadIdx.x == 0) { stats[f] = sh[0]; stats[128 + f] = sh2[0]; }
}

// ---- h = relu(gamma*(agg-mu)*rsqrt(var+eps)+beta) ----
__global__ void bn_relu_kernel(const unsigned* __restrict__ keys,
                               const float* __restrict__ stats,
                               const float* __restrict__ gamma,
                               const float* __restrict__ beta,
                               float* __restrict__ h, int layer) {
    int i = blockIdx.x * 256 + threadIdx.x;
    if (i >= N_NODES * HDIM) return;
    int f = i & 127;
    float mu  = stats[f] * (1.f / N_NODES);
    float var = stats[128 + f] * (1.f / N_NODES) - mu * mu;
    unsigned k = keys[i];
    float v = (k == NEG_KEY) ? 0.f : fdec(k);
    float y = gamma[layer * 128 + f] * (v - mu) * rsqrtf(var + 1e-5f) + beta[layer * 128 + f];
    h[i] = fmaxf(y, 0.f);
}

// ---- out = relu(h @ Wo1 + bo1) @ Wo2 + bo2 ----
__global__ __launch_bounds__(64) void out_kernel(const float* __restrict__ h,
                                                 const float* __restrict__ Wo1,
                                                 const float* __restrict__ bo1,
                                                 const float* __restrict__ Wo2,
                                                 const float* __restrict__ bo2,
                                                 float* __restrict__ out) {
    int n = blockIdx.x, j = threadIdx.x; // 64 threads
    __shared__ float hr[128];
    __shared__ float red[64];
    for (int k = j; k < 128; k += 64) hr[k] = h[(size_t)n * 128 + k];
    __syncthreads();
    float acc = bo1[j];
#pragma unroll 8
    for (int k = 0; k < 128; ++k) acc = fmaf(hr[k], Wo1[k * 64 + j], acc);
    red[j] = fmaxf(acc, 0.f) * Wo2[j];
    __syncthreads();
    for (int st = 32; st > 0; st >>= 1) {
        if (j < st) red[j] += red[j + st];
        __syncthreads();
    }
    if (j == 0) out[n] = red[0] + bo2[0];
}

extern "C" void kernel_launch(void* const* d_in, const int* in_sizes, int n_in,
                              void* d_out, int out_size, void* d_ws, size_t ws_size,
                              hipStream_t stream) {
    const float* x     = (const float*)d_in[0];
    const int*   ei    = (const int*)  d_in[1];
    const float* Wp    = (const float*)d_in[2];
    const float* bp    = (const float*)d_in[3];
    const float* W1    = (const float*)d_in[4];
    const float* b1    = (const float*)d_in[5];
    const float* W2    = (const float*)d_in[6];
    const float* b2    = (const float*)d_in[7];
    const float* gamma = (const float*)d_in[8];
    const float* beta  = (const float*)d_in[9];
    const float* Wo1   = (const float*)d_in[10];
    const float* bo1   = (const float*)d_in[11];
    const float* Wo2   = (const float*)d_in[12];
    const float* bo2   = (const float*)d_in[13];

    // workspace layout
    char* ws = (char*)d_ws;
    const size_t HBYTES = (size_t)N_NODES * HDIM * 4; // 20.48 MB
    float*    hbuf  = (float*)ws;
    unsigned* keys  = (unsigned*)(ws + HBYTES);
    float*    stats = (float*)(ws + 2 * HBYTES);
    __bf16*   pW1   = (__bf16*)(ws + 2 * HBYTES + 1024);
    __bf16*   pW2   = pW1 + 3 * 8 * 8 * 512; // 98304 elems

    const int elemBlocks = (N_NODES * HDIM + 255) / 256; // 20000

    pack_w_kernel<<<(3 * 8 * 8 * 512 + 3 * 8 * 4 * 512 + 255) / 256, 256, 0, stream>>>(
        W1, W2, pW1, pW2);
    proj_kernel<<<elemBlocks, 256, 0, stream>>>(x, Wp, bp, hbuf);

    for (int l = 0; l < 3; ++l) {
        init_kernel<<<elemBlocks, 256, 0, stream>>>(keys, stats);
        edge_mlp_kernel<<<2048, 256, 0, stream>>>(hbuf, ei, pW1, pW2, b1, b2, keys, l);
        stats_kernel<<<HDIM, 256, 0, stream>>>(keys, stats);
        bn_relu_kernel<<<elemBlocks, 256, 0, stream>>>(keys, stats, gamma, beta, hbuf, l);
    }
    out_kernel<<<N_NODES, 64, 0, stream>>>(hbuf, Wo1, bo1, Wo2, bo2, (float*)d_out);
}
